// Gru_cond_layer_58737972740695
// MI455X (gfx1250) — matrix operand order repaired and checked by reading, not compiled, not run on hardware
//
#include <hip/hip_runtime.h>
#include <hip/hip_bf16.h>
#include <math.h>

// ---------------- problem constants ----------------
#define T_   48
#define B_   16
#define H_   16
#define W_   48
#define HW_  768          // H*W
#define D_   684
#define N_   256
#define M_   256
#define A_   512
#define CQ_  512
#define KQ_  121          // 11*11
#define NKT_UA 22         // ceil(684/32) -> 704 padded K

// output offsets (float elements), concatenated in reference return order
#define OUT_H2     0
#define OUT_CTS    196608     // 48*16*256
#define OUT_ALPHAS 721920     // + 48*16*684
#define OUT_APAST  1311744    // + 48*16*768

// workspace offsets (bytes)
#define OFF_H      0u
#define OFF_H1     16384u
#define OFF_WAH1   32768u
#define OFF_AP     65536u
#define OFF_ALPHA  114688u
#define OFF_E      163840u
#define OFF_CT     212992u
#define OFF_SZ     258048u
#define OFF_SR     1044480u
#define OFF_SH     1830912u
#define OFF_QUF    2617344u     // bf16 frag-major [4][32][32][16]
#define OFF_UAWF   2748416u     // bf16 frag-major [22][32][32][16]
#define OFF_UACTX  3469312u     // f32 [12288][512]

typedef __attribute__((ext_vector_type(16))) __bf16 v16bf;
typedef __attribute__((ext_vector_type(8)))  float  v8f;

__device__ __forceinline__ unsigned short f32_to_bf16(float f) {
  unsigned int u = __float_as_uint(f);
  unsigned int lsb = (u >> 16) & 1u;
  u += 0x7fffu + lsb;                 // round-to-nearest-even
  return (unsigned short)(u >> 16);
}

// A-fragment slot->local-K map (16-bit A 16x32): K = 16*half + s
__device__ __forceinline__ int a_klocal(int s, int half) {
  return 16 * half + s;
}
// B-fragment slot->local-K map (16-bit B 32x16): K = 16*half + s

// ---------------- init: h = init_state, alpha_past = 0 ----------------
__global__ void init_state_kernel(const float* __restrict__ init_state,
                                  float* __restrict__ h, float* __restrict__ ap) {
  int idx = blockIdx.x * 256 + threadIdx.x;      // 12288 threads
  if (idx < B_ * N_) h[idx] = init_state[idx];
  if (idx < B_ * HW_) ap[idx] = 0.f;
}

// ---------------- embedding projections sz/sr/sh ----------------
__global__ __launch_bounds__(256) void szrh_kernel(
    const float* __restrict__ emb,
    const float* __restrict__ Wz, const float* __restrict__ bz,
    const float* __restrict__ Wr, const float* __restrict__ br,
    const float* __restrict__ Wh, const float* __restrict__ bh,
    float* __restrict__ sz, float* __restrict__ sr, float* __restrict__ sh) {
  int idx = blockIdx.x * 256 + threadIdx.x;      // 768*256
  int tb = idx >> 8, n = idx & 255;
  const float* er = emb + tb * M_;
  const float* wz = Wz + n * M_;
  const float* wr = Wr + n * M_;
  const float* wh = Wh + n * M_;
  float az = 0.f, ar = 0.f, ah = 0.f;
  for (int k = 0; k < M_; ++k) {
    float ev = er[k];
    az += wz[k] * ev; ar += wr[k] * ev; ah += wh[k] * ev;
  }
  sz[idx] = az + bz[n]; sr[idx] = ar + br[n]; sh[idx] = ah + bh[n];
}

// ------- QU = Uf_w @ Q_w.reshape(512,121), packed B-fragment-major bf16 -------
__global__ __launch_bounds__(256) void quf_kernel(
    const float* __restrict__ Uf_w, const float* __restrict__ Q_w,
    unsigned short* __restrict__ quf) {
  int flat = blockIdx.x * 256 + threadIdx.x;     // 4*32*32*16 = 65536
  int s = flat & 15;
  int lane = (flat >> 4) & 31;
  int nt = (flat >> 9) & 31;
  int kt = flat >> 14;
  int k = kt * 32 + 16 * (lane >> 4) + s;        // B-frag K
  int a = nt * 16 + (lane & 15);                 // B-frag N
  float val = 0.f;
  if (k < KQ_) {
    const float* ur = Uf_w + a * CQ_;
    float acc = 0.f;
    for (int c = 0; c < CQ_; ++c) acc += ur[c] * Q_w[c * KQ_ + k];
    val = acc;
  }
  quf[flat] = f32_to_bf16(val);
}

// ------- Ua_w^T packed B-fragment-major bf16, K padded 684->704 -------
__global__ __launch_bounds__(256) void uawf_kernel(
    const float* __restrict__ Ua_w, unsigned short* __restrict__ uawf) {
  int flat = blockIdx.x * 256 + threadIdx.x;     // 22*32*32*16 = 360448
  if (flat >= NKT_UA * 32 * 32 * 16) return;
  int s = flat & 15;
  int lane = (flat >> 4) & 31;
  int nt = (flat >> 9) & 31;
  int kt = flat >> 14;
  int k = kt * 32 + 16 * (lane >> 4) + s;
  int n = nt * 16 + (lane & 15);
  uawf[flat] = f32_to_bf16((k < D_) ? Ua_w[n * D_ + k] : 0.f);
}

// ------- Ua_ctx[(hw*16+b)][a] = context[b,:,hw] . Ua_w[a,:] + Ua_b  (WMMA) -------
__global__ __launch_bounds__(256) void uactx_kernel(
    const float* __restrict__ context, const float* __restrict__ Ua_b,
    const unsigned short* __restrict__ uawf, float* __restrict__ ua_ctx) {
  __shared__ __align__(32) unsigned short Af[32][16];   // one 16x32 A chunk
  const int tid = threadIdx.x;
  const int hw = blockIdx.x;
  const int lane = tid & 31;
  const int wave = tid >> 5;                            // 8 waves (wave32)
  v8f acc[4];
  for (int j = 0; j < 4; ++j) acc[j] = (v8f){0.f,0.f,0.f,0.f,0.f,0.f,0.f,0.f};
  for (int kt = 0; kt < NKT_UA; ++kt) {
    for (int it = 0; it < 2; ++it) {                    // 512 elems / 256 thr
      int flat = tid + it * 256;
      int s = flat & 15;
      int ln = flat >> 4;
      int hf = ln >> 4;
      int m = ln & 15;                                  // batch
      int d = kt * 32 + a_klocal(s, hf);
      float val = (d < D_) ? context[(m * D_ + d) * HW_ + hw] : 0.f;
      Af[ln][s] = f32_to_bf16(val);
    }
    __syncthreads();
    v16bf a = *reinterpret_cast<const v16bf*>(&Af[lane][0]);
    for (int j = 0; j < 4; ++j) {
      int nt = wave * 4 + j;
      v16bf b = *reinterpret_cast<const v16bf*>(
          &uawf[((kt * 32 + nt) * 32 + lane) * 16]);
      acc[j] = __builtin_amdgcn_wmma_f32_16x16x32_bf16(
          false, a, false, b, (short)0, acc[j], false, false);
    }
    __syncthreads();
  }
  const int half = lane >> 4, ncol = lane & 15;
  for (int j = 0; j < 4; ++j) {
    int n = (wave * 4 + j) * 16 + ncol;
    float bias = Ua_b[n];
    for (int v = 0; v < 8; ++v) {
      int m = v + 8 * half;
      ua_ctx[(size_t)(hw * 16 + m) * A_ + n] = acc[j][v] + bias;
    }
  }
}

// ---------------- GRU 1 ----------------
__global__ __launch_bounds__(256) void gru1_kernel(
    const float* __restrict__ h, const float* __restrict__ sz,
    const float* __restrict__ sr, const float* __restrict__ sh,
    const float* __restrict__ mask,
    const float* __restrict__ Uhz, const float* __restrict__ Uhr,
    const float* __restrict__ Uhh, float* __restrict__ h1, int t) {
  __shared__ float hs[N_];
  int b = blockIdx.x, n = threadIdx.x;
  hs[n] = h[b * N_ + n];
  __syncthreads();
  const float* wz = Uhz + n * N_;
  const float* wr = Uhr + n * N_;
  const float* wh = Uhh + n * N_;
  float az = 0.f, ar = 0.f, ah = 0.f;
  for (int k = 0; k < N_; ++k) {
    float hv = hs[k];
    az += wz[k] * hv; ar += wr[k] * hv; ah += wh[k] * hv;
  }
  int tb = t * B_ + b;
  float z1 = 1.f / (1.f + expf(-(az + sz[tb * N_ + n])));
  float r1 = 1.f / (1.f + expf(-(ar + sr[tb * N_ + n])));
  float hp = tanhf(ah * r1 + sh[tb * N_ + n]);
  float ho = hs[n];
  float hn = z1 * ho + (1.f - z1) * hp;
  float m = mask[tb];
  h1[b * N_ + n] = m * hn + (1.f - m) * ho;
}

// ---------------- Wa_h1 = h1 @ Wa_w^T ----------------
__global__ __launch_bounds__(256) void wa_kernel(
    const float* __restrict__ h1, const float* __restrict__ Wa_w,
    float* __restrict__ wah1) {
  __shared__ float hs[N_];
  int b = blockIdx.x >> 1;
  int a = ((blockIdx.x & 1) << 8) + threadIdx.x;
  for (int i = threadIdx.x; i < N_; i += 256) hs[i] = h1[b * N_ + i];
  __syncthreads();
  const float* w = Wa_w + a * N_;
  float s = 0.f;
  for (int k = 0; k < N_; ++k) s += w[k] * hs[k];
  wah1[b * A_ + a] = s;
}

// ------- fused coverage-conv GEMM + attention energy (the big WMMA kernel) -------
// one workgroup per (h,w); rows = 16 batches; e[hw*16+b] = va . tanh(pre) + va_b
__global__ __launch_bounds__(256) void cover_att_kernel(
    const float* __restrict__ ap, const unsigned short* __restrict__ quf,
    const float* __restrict__ ua_ctx, const float* __restrict__ wah1,
    const float* __restrict__ Uf_b, const float* __restrict__ va_w,
    const float* __restrict__ va_b, float* __restrict__ e) {
  __shared__ __align__(32) unsigned short Apf[4][32][16];  // im2col A frags, bf16
  __shared__ float esum[16];
  const int tid = threadIdx.x;
  const int hw = blockIdx.x;
  const int hh = hw / W_;
  const int ww = hw % W_;
  if (tid < 16) esum[tid] = 0.f;
  // im2col: patch[b, k] = alpha_past[b, hh+i-5, ww+j-5], k = i*11+j, pad to 128
  for (int it = 0; it < 8; ++it) {
    int flat = tid + it * 256;             // 2048 elems
    int s = flat & 15;
    int lane = (flat >> 4) & 31;
    int kt = flat >> 9;
    int half = lane >> 4;
    int m = lane & 15;                     // batch
    int k = kt * 32 + a_klocal(s, half);
    float val = 0.f;
    if (k < KQ_) {
      int i = k / 11, j = k - 11 * i;
      int rr = hh + i - 5, cc = ww + j - 5;
      if (rr >= 0 && rr < H_ && cc >= 0 && cc < W_)
        val = ap[m * HW_ + rr * W_ + cc];
    }
    Apf[kt][lane][s] = f32_to_bf16(val);
  }
  __syncthreads();

  const int lane = tid & 31;
  const int wave = tid >> 5;
  v8f acc[4];
  for (int j = 0; j < 4; ++j) {
    v8f c = (v8f){0.f,0.f,0.f,0.f,0.f,0.f,0.f,0.f};
    int nt = wave * 4 + j;
    for (int kt = 0; kt < 4; ++kt) {
      v16bf a = *reinterpret_cast<const v16bf*>(&Apf[kt][lane][0]);
      v16bf b = *reinterpret_cast<const v16bf*>(
          &quf[((kt * 32 + nt) * 32 + lane) * 16]);
      c = __builtin_amdgcn_wmma_f32_16x16x32_bf16(
          false, a, false, b, (short)0, c, false, false);
    }
    acc[j] = c;
  }
  // epilogue: pre = cover + Uf_b + Ua_ctx + Wa_h1 ; e += va * tanh(pre)
  float rsum[8];
  for (int v = 0; v < 8; ++v) rsum[v] = 0.f;
  const int half = lane >> 4;
  const int ncol = lane & 15;
  for (int j = 0; j < 4; ++j) {
    int n = (wave * 4 + j) * 16 + ncol;
    float ufb = Uf_b[n];
    float van = va_w[n];
    for (int v = 0; v < 8; ++v) {
      int m = v + 8 * half;
      float pre = acc[j][v] + ufb +
                  ua_ctx[(size_t)(hw * 16 + m) * A_ + n] + wah1[m * A_ + n];
      rsum[v] += tanhf(pre) * van;
    }
  }
  for (int v = 0; v < 8; ++v) {             // reduce over 16 columns (half-wave)
    float s = rsum[v];
    s += __shfl_xor(s, 1, 32);
    s += __shfl_xor(s, 2, 32);
    s += __shfl_xor(s, 4, 32);
    s += __shfl_xor(s, 8, 32);
    if (ncol == 0) atomicAdd(&esum[v + 8 * half], s);
  }
  __syncthreads();
  if (tid < 16) e[hw * 16 + tid] = esum[tid] + va_b[0];
}

// ---------------- masked softmax over H*W, alpha_past update ----------------
__global__ __launch_bounds__(256) void softmax_kernel(
    const float* __restrict__ e, const float* __restrict__ cmask,
    float* __restrict__ alpha, float* __restrict__ ap,
    float* __restrict__ out, int t) {
  __shared__ float red[256];
  const int b = blockIdx.x, tid = threadIdx.x;
  float mx = -1e30f;
  for (int hw = tid; hw < HW_; hw += 256) mx = fmaxf(mx, e[hw * 16 + b]);
  red[tid] = mx; __syncthreads();
  for (int s = 128; s > 0; s >>= 1) {
    if (tid < s) red[tid] = fmaxf(red[tid], red[tid + s]);
    __syncthreads();
  }
  mx = red[0]; __syncthreads();
  float sum = 0.f;
  for (int hw = tid; hw < HW_; hw += 256)
    sum += expf(e[hw * 16 + b] - mx) * cmask[b * HW_ + hw];
  red[tid] = sum; __syncthreads();
  for (int s = 128; s > 0; s >>= 1) {
    if (tid < s) red[tid] += red[tid + s];
    __syncthreads();
  }
  float inv = 1.f / red[0];
  float* alphas_out = out + OUT_ALPHAS + (size_t)t * B_ * HW_;
  float* appast_out = out + OUT_APAST + (size_t)t * B_ * HW_;
  for (int hw = tid; hw < HW_; hw += 256) {
    float al = expf(e[hw * 16 + b] - mx) * cmask[b * HW_ + hw] * inv;
    alpha[b * HW_ + hw] = al;
    float apn = ap[b * HW_ + hw] + al;
    ap[b * HW_ + hw] = apn;
    alphas_out[b * HW_ + hw] = al;
    appast_out[b * HW_ + hw] = apn;
  }
}

// ---------------- ct[b,d] = sum_hw context[b,d,hw] * alpha[b,hw] ----------------
__global__ __launch_bounds__(256) void ct_kernel(
    const float* __restrict__ context, const float* __restrict__ alpha,
    float* __restrict__ ct, float* __restrict__ out, int t) {
  int idx = blockIdx.x * 256 + threadIdx.x;
  if (idx >= B_ * D_) return;
  int b = idx / D_, d = idx - b * D_;
  const float* crow = context + (size_t)(b * D_ + d) * HW_;
  const float* arow = alpha + b * HW_;
  float s = 0.f;
  for (int hw = 0; hw < HW_; ++hw) s += crow[hw] * arow[hw];
  ct[idx] = s;
  out[OUT_CTS + (size_t)t * B_ * D_ + idx] = s;
}

// ---------------- GRU 2 ----------------
__global__ __launch_bounds__(256) void gru2_kernel(
    const float* __restrict__ ct, const float* __restrict__ h1,
    const float* __restrict__ mask,
    const float* __restrict__ Wcz, const float* __restrict__ Wcr,
    const float* __restrict__ Wch,
    const float* __restrict__ Uhz2, const float* __restrict__ bz2,
    const float* __restrict__ Uhr2, const float* __restrict__ br2,
    const float* __restrict__ Uhh2, const float* __restrict__ bh2,
    float* __restrict__ h, float* __restrict__ out, int t) {
  __shared__ float cts[D_];
  __shared__ float h1s[N_];
  int b = blockIdx.x, n = threadIdx.x;
  for (int i = n; i < D_; i += 256) cts[i] = ct[b * D_ + i];
  h1s[n] = h1[b * N_ + n];
  __syncthreads();
  const float* wz = Wcz + n * D_;
  const float* wr = Wcr + n * D_;
  const float* wh = Wch + n * D_;
  float cz = 0.f, cr = 0.f, ch = 0.f;
  for (int k = 0; k < D_; ++k) {
    float c = cts[k];
    cz += wz[k] * c; cr += wr[k] * c; ch += wh[k] * c;
  }
  const float* vz = Uhz2 + n * N_;
  const float* vr = Uhr2 + n * N_;
  const float* vh = Uhh2 + n * N_;
  float uz = 0.f, ur = 0.f, uh = 0.f;
  for (int k = 0; k < N_; ++k) {
    float hv = h1s[k];
    uz += vz[k] * hv; ur += vr[k] * hv; uh += vh[k] * hv;
  }
  float z2 = 1.f / (1.f + expf(-(cz + uz + bz2[n])));
  float r2 = 1.f / (1.f + expf(-(cr + ur + br2[n])));
  float hp = tanhf(ch + (uh + bh2[n]) * r2);
  float h1v = h1s[n];
  float h2 = z2 * h1v + (1.f - z2) * hp;
  float m = mask[t * B_ + b];
  h2 = m * h2 + (1.f - m) * h1v;
  h[b * N_ + n] = h2;
  out[(size_t)(t * B_ + b) * N_ + n] = h2;
}

// ---------------- launch ----------------
extern "C" void kernel_launch(void* const* d_in, const int* in_sizes, int n_in,
                              void* d_out, int out_size, void* d_ws, size_t ws_size,
                              hipStream_t stream) {
  (void)in_sizes; (void)n_in; (void)out_size; (void)ws_size;
  const float* embedding  = (const float*)d_in[0];
  const float* mask       = (const float*)d_in[1];
  const float* context    = (const float*)d_in[2];
  const float* cmask      = (const float*)d_in[3];
  const float* init_state = (const float*)d_in[4];
  const float* Ua_w  = (const float*)d_in[5];
  const float* Ua_b  = (const float*)d_in[6];
  const float* Wa_w  = (const float*)d_in[7];
  const float* Q_w   = (const float*)d_in[8];
  const float* Uf_w  = (const float*)d_in[9];
  const float* Uf_b  = (const float*)d_in[10];
  const float* va_w  = (const float*)d_in[11];
  const float* va_b  = (const float*)d_in[12];
  const float* Wyz_w = (const float*)d_in[13]; const float* Wyz_b = (const float*)d_in[14];
  const float* Wyr_w = (const float*)d_in[15]; const float* Wyr_b = (const float*)d_in[16];
  const float* Wyh_w = (const float*)d_in[17]; const float* Wyh_b = (const float*)d_in[18];
  const float* Uhz_w = (const float*)d_in[19];
  const float* Uhr_w = (const float*)d_in[20];
  const float* Uhh_w = (const float*)d_in[21];
  const float* Wcz_w = (const float*)d_in[22];
  const float* Wcr_w = (const float*)d_in[23];
  const float* Wch_w = (const float*)d_in[24];
  const float* Uhz2_w = (const float*)d_in[25]; const float* Uhz2_b = (const float*)d_in[26];
  const float* Uhr2_w = (const float*)d_in[27]; const float* Uhr2_b = (const float*)d_in[28];
  const float* Uhh2_w = (const float*)d_in[29]; const float* Uhh2_b = (const float*)d_in[30];

  char* ws = (char*)d_ws;
  float* h      = (float*)(ws + OFF_H);
  float* h1     = (float*)(ws + OFF_H1);
  float* wah1   = (float*)(ws + OFF_WAH1);
  float* ap     = (float*)(ws + OFF_AP);
  float* alpha  = (float*)(ws + OFF_ALPHA);
  float* e      = (float*)(ws + OFF_E);
  float* ctv    = (float*)(ws + OFF_CT);
  float* sz     = (float*)(ws + OFF_SZ);
  float* sr     = (float*)(ws + OFF_SR);
  float* sh     = (float*)(ws + OFF_SH);
  unsigned short* quf  = (unsigned short*)(ws + OFF_QUF);
  unsigned short* uawf = (unsigned short*)(ws + OFF_UAWF);
  float* ua_ctx = (float*)(ws + OFF_UACTX);
  float* out = (float*)d_out;

  dim3 blk(256);
  init_state_kernel<<<48, blk, 0, stream>>>(init_state, h, ap);
  szrh_kernel<<<768, blk, 0, stream>>>(embedding, Wyz_w, Wyz_b, Wyr_w, Wyr_b,
                                       Wyh_w, Wyh_b, sz, sr, sh);
  quf_kernel<<<256, blk, 0, stream>>>(Uf_w, Q_w, quf);
  uawf_kernel<<<1408, blk, 0, stream>>>(Ua_w, uawf);
  uactx_kernel<<<768, blk, 0, stream>>>(context, Ua_b, uawf, ua_ctx);

  for (int t = 0; t < T_; ++t) {
    gru1_kernel<<<16, blk, 0, stream>>>(h, sz, sr, sh, mask,
                                        Uhz_w, Uhr_w, Uhh_w, h1, t);
    wa_kernel<<<32, blk, 0, stream>>>(h1, Wa_w, wah1);
    cover_att_kernel<<<768, blk, 0, stream>>>(ap, quf, ua_ctx, wah1,
                                              Uf_b, va_w, va_b, e);
    softmax_kernel<<<16, blk, 0, stream>>>(e, cmask, alpha, ap, out, t);
    ct_kernel<<<43, blk, 0, stream>>>(context, alpha, ctv, out, t);
    gru2_kernel<<<16, blk, 0, stream>>>(ctv, h1, mask, Wcz_w, Wcr_w, Wch_w,
                                        Uhz2_w, Uhz2_b, Uhr2_w, Uhr2_b,
                                        Uhh2_w, Uhh2_b, h, out, t);
  }
}
